// HeteroGraph_33629593928253
// MI455X (gfx1250) — matrix operand order repaired and checked by reading, not compiled, not run on hardware
//
#include <hip/hip_runtime.h>

typedef __attribute__((ext_vector_type(16))) _Float16 v16h;
typedef __attribute__((ext_vector_type(8)))  float    v8f;

// ---------------------------------------------------------------- utilities

__global__ void zero_f32(float* __restrict__ p, long long n) {
  long long i  = (long long)blockIdx.x * blockDim.x + threadIdx.x;
  long long st = (long long)gridDim.x * blockDim.x;
  for (; i < n; i += st) p[i] = 0.f;
}

// Pack f32 W[K=128][N=128] row-major into WMMA B fragments (f16).
// Layout: [tile = tk*8+tn][lane][16 halfs]; lane holds col n = tn*16+(lane&15),
// K = (lane>>4)*8 + j for j<8, 16 + (lane>>4)*8 + (j-8) for j>=8 within the 32-wide K tile.
__global__ void pack_w(const float* __restrict__ W, _Float16* __restrict__ Bp) {
  int tile = blockIdx.x;                 // 32 tiles
  int tk = tile >> 3, tn = tile & 7;
  int lane = threadIdx.x;                // 32 threads
  int n  = tn * 16 + (lane & 15);
  int kb = tk * 32 + ((lane >> 4) << 3);
  _Float16* dst = Bp + ((size_t)tile * 32 + lane) * 16;
#pragma unroll
  for (int j = 0; j < 16; ++j)
    dst[j] = (_Float16)W[(size_t)(kb + j + ((j >> 3) << 3)) * 128 + n];
}

// Stage the 32 KB packed-weight block into LDS with gfx1250 async
// global->LDS B128 transfers (no VGPR round-trip). Each of the 256 threads
// moves 128 contiguous bytes; the instruction offset advances both the
// global source and the LDS destination. Caller must __syncthreads() after.
__device__ __forceinline__ void stage_B_async(const _Float16* __restrict__ Bp,
                                              _Float16* Bsh) {
  unsigned ldsAddr = (unsigned)(uintptr_t)Bsh + threadIdx.x * 128u;  // addr[31:0] = LDS offset
  unsigned gOff    = threadIdx.x * 128u;
  asm volatile(
      "global_load_async_to_lds_b128 %0, %1, %2 offset:0\n\t"
      "global_load_async_to_lds_b128 %0, %1, %2 offset:16\n\t"
      "global_load_async_to_lds_b128 %0, %1, %2 offset:32\n\t"
      "global_load_async_to_lds_b128 %0, %1, %2 offset:48\n\t"
      "global_load_async_to_lds_b128 %0, %1, %2 offset:64\n\t"
      "global_load_async_to_lds_b128 %0, %1, %2 offset:80\n\t"
      "global_load_async_to_lds_b128 %0, %1, %2 offset:96\n\t"
      "global_load_async_to_lds_b128 %0, %1, %2 offset:112\n\t"
      "s_wait_asynccnt 0"
      :: "v"(ldsAddr), "v"(gOff), "s"(Bp)
      : "memory");
}

// Load 4 A-fragments (16 rows x K=128, f32 -> f16) for one wave.
template <bool RELU>
__device__ __forceinline__ void load_a_frags(const float* __restrict__ X,
                                             int row0, int nrows, int lane,
                                             v16h a[4]) {
  int m = lane & 15;
  int khalf = (lane >> 4) << 3;          // 0 or 8
  int r = row0 + m; if (r >= nrows) r = nrows - 1;
  const float* xr = X + (size_t)r * 128;
#pragma unroll
  for (int tk = 0; tk < 4; ++tk) {
    const float4* p0 = (const float4*)(xr + tk * 32 + khalf);
    const float4* p1 = (const float4*)(xr + tk * 32 + khalf + 16);
    float4 u = p0[0], v = p0[1], w = p1[0], z = p1[1];
    if (RELU) {
      u.x = fmaxf(u.x, 0.f); u.y = fmaxf(u.y, 0.f); u.z = fmaxf(u.z, 0.f); u.w = fmaxf(u.w, 0.f);
      v.x = fmaxf(v.x, 0.f); v.y = fmaxf(v.y, 0.f); v.z = fmaxf(v.z, 0.f); v.w = fmaxf(v.w, 0.f);
      w.x = fmaxf(w.x, 0.f); w.y = fmaxf(w.y, 0.f); w.z = fmaxf(w.z, 0.f); w.w = fmaxf(w.w, 0.f);
      z.x = fmaxf(z.x, 0.f); z.y = fmaxf(z.y, 0.f); z.z = fmaxf(z.z, 0.f); z.w = fmaxf(z.w, 0.f);
    }
    a[tk][0]=(_Float16)u.x; a[tk][1]=(_Float16)u.y; a[tk][2]=(_Float16)u.z; a[tk][3]=(_Float16)u.w;
    a[tk][4]=(_Float16)v.x; a[tk][5]=(_Float16)v.y; a[tk][6]=(_Float16)v.z; a[tk][7]=(_Float16)v.w;
    a[tk][8]=(_Float16)w.x; a[tk][9]=(_Float16)w.y; a[tk][10]=(_Float16)w.z; a[tk][11]=(_Float16)w.w;
    a[tk][12]=(_Float16)z.x; a[tk][13]=(_Float16)z.y; a[tk][14]=(_Float16)z.z; a[tk][15]=(_Float16)z.w;
  }
}

// ------------------------------------------------ WMMA GEMM: Y = X@W + bias
// B (32 KB packed weights) is staged through LDS once per 256-thread block.
__global__ void __launch_bounds__(256)
gemm_bias(const float* __restrict__ X, const _Float16* __restrict__ Bp,
          const float* __restrict__ bias, float* __restrict__ Y, int nrows) {
  __shared__ __align__(32) _Float16 Bsh[16384];
  stage_B_async(Bp, Bsh);
  __syncthreads();
  int lane = threadIdx.x & 31;
  int row0 = (blockIdx.x * 8 + (threadIdx.x >> 5)) << 4;
  if (row0 >= nrows) return;             // wave-uniform: EXEC all-ones for WMMA
  v16h a[4];
  load_a_frags<false>(X, row0, nrows, lane, a);
  int mlo = lane & 15, hi = lane >> 4;
  bool full = (row0 + 16 <= nrows);      // wave-uniform fast path
#pragma unroll
  for (int tn = 0; tn < 8; ++tn) {
    float bv = bias[tn * 16 + mlo];
    v8f c;
#pragma unroll
    for (int i = 0; i < 8; ++i) c[i] = bv;
#pragma unroll
    for (int tk = 0; tk < 4; ++tk) {
      v16h b = *(const v16h*)(Bsh + (((size_t)tk * 8 + tn) * 32 + lane) * 16);
      c = __builtin_amdgcn_wmma_f32_16x16x32_f16(false, a[tk], false, b,
                                                 (short)0, c, false, false);
    }
    int n = tn * 16 + mlo;
    if (full) {
      float* yb = Y + (size_t)(row0 + hi * 8) * 128 + n;
#pragma unroll
      for (int r = 0; r < 8; ++r) yb[(size_t)r * 128] = c[r];
    } else {
#pragma unroll
      for (int r = 0; r < 8; ++r) {
        int rr = row0 + r + hi * 8;
        if (rr < nrows) Y[(size_t)rr * 128 + n] = c[r];
      }
    }
  }
}

// WMMA GEMM fused semantic-attention score:
//   score += sum_rows q . tanh(relu(X)@kW + kb)
__global__ void __launch_bounds__(256)
gemm_score(const float* __restrict__ X, const _Float16* __restrict__ Bp,
           const float* __restrict__ kb, const float* __restrict__ q,
           float* __restrict__ score, int nrows) {
  __shared__ __align__(32) _Float16 Bsh[16384];
  stage_B_async(Bp, Bsh);
  __syncthreads();
  int lane = threadIdx.x & 31;
  int row0 = (blockIdx.x * 8 + (threadIdx.x >> 5)) << 4;
  if (row0 >= nrows) return;
  v16h a[4];
  load_a_frags<true>(X, row0, nrows, lane, a);   // relu fused into A-load
  int mlo = lane & 15, hi = lane >> 4;
  bool full = (row0 + 16 <= nrows);
  float acc = 0.f;
#pragma unroll
  for (int tn = 0; tn < 8; ++tn) {
    float bv = kb[tn * 16 + mlo];
    float qv = q[tn * 16 + mlo];
    v8f c;
#pragma unroll
    for (int i = 0; i < 8; ++i) c[i] = bv;
#pragma unroll
    for (int tk = 0; tk < 4; ++tk) {
      v16h b = *(const v16h*)(Bsh + (((size_t)tk * 8 + tn) * 32 + lane) * 16);
      c = __builtin_amdgcn_wmma_f32_16x16x32_f16(false, a[tk], false, b,
                                                 (short)0, c, false, false);
    }
    if (full) {
#pragma unroll
      for (int r = 0; r < 8; ++r) acc += tanhf(c[r]) * qv;
    } else {
#pragma unroll
      for (int r = 0; r < 8; ++r)
        if (row0 + r + hi * 8 < nrows) acc += tanhf(c[r]) * qv;
    }
  }
#pragma unroll
  for (int o = 16; o; o >>= 1) acc += __shfl_xor(acc, o, 32);
  if (lane == 0) atomicAdd(score, acc);
}

// --------------------------------------------------------- small elementwise

// y[n, :128] = x[n, :in] @ W[in,128] + b   (block = one node, 128 threads)
__global__ void encoder(const float* __restrict__ x, const float* __restrict__ W,
                        const float* __restrict__ b, float* __restrict__ y,
                        int n, int indim) {
  int node = blockIdx.x, f = threadIdx.x;
  if (node >= n) return;
  float acc = b[f];
  for (int i = 0; i < indim; ++i)
    acc += x[(size_t)node * indim + i] * W[i * 128 + f];
  y[(size_t)node * 128 + f] = acc;
}

// Up to 8 attention dots per node type in one pass over X (wave per row).
struct DotArgs { const float* v[8]; float* y[8]; };

__global__ void rowdot_multi(const float* __restrict__ X, DotArgs da, int cnt, int n) {
  int lane = threadIdx.x & 31;
  int row = blockIdx.x * 8 + (threadIdx.x >> 5);
  if (row >= n) return;
  float4 a = ((const float4*)(X + (size_t)row * 128))[lane];
  for (int j = 0; j < cnt; ++j) {
    float4 b = ((const float4*)da.v[j])[lane];
    float s = a.x * b.x + a.y * b.y + a.z * b.z + a.w * b.w;
#pragma unroll
    for (int o = 16; o; o >>= 1) s += __shfl_xor(s, o, 32);
    if (lane == 0) da.y[j][row] = s;
  }
}

// ex[e] = exp(leaky_relu(asrc[src]+adst[dst])); denom[dst] += ex
__global__ void edge_pass1(const int* __restrict__ src, const int* __restrict__ dst,
                           const float* __restrict__ asrc, const float* __restrict__ adst,
                           float* __restrict__ ex, float* __restrict__ denom, int E) {
  int e = blockIdx.x * 256 + threadIdx.x;
  if (e >= E) return;
  float a = asrc[src[e]] + adst[dst[e]];
  a = a > 0.f ? a : 0.2f * a;
  float v = __expf(a);
  ex[e] = v;
  atomicAdd(&denom[dst[e]], v);
}

// agg[dst,:] += Xs[src,:] * ex[e]/(denom[dst]+eps)   (2 edges per block)
__global__ void edge_pass2(const int* __restrict__ src, const int* __restrict__ dst,
                           const float* __restrict__ ex, const float* __restrict__ denom,
                           const float* __restrict__ Xs, float* __restrict__ agg, int E) {
  int e = blockIdx.x * 2 + (threadIdx.x >> 7);
  int f = threadIdx.x & 127;
  if (e >= E) return;
  int s = src[e], d = dst[e];
  float w = ex[e] / (denom[d] + 1e-16f);
  atomicAdd(&agg[(size_t)d * 128 + f], Xs[(size_t)s * 128 + f] * w);
}

__global__ void softmax_w(const float* __restrict__ score, float* __restrict__ w,
                          int M, float invN) {
  if (threadIdx.x != 0 || blockIdx.x != 0) return;
  float s[4]; float mx = -1e30f;
  for (int m = 0; m < M; ++m) { s[m] = score[m] * invN; mx = fmaxf(mx, s[m]); }
  float den = 0.f;
  for (int m = 0; m < M; ++m) { s[m] = __expf(s[m] - mx); den += s[m]; }
  for (int m = 0; m < M; ++m) w[m] = s[m] / den;
}

// out[row,:] = LayerNorm(ELU( sum_m w[m]*relu(agg_m[row,:]) ))   (wave per row)
__global__ void combine_elu_ln(const float* __restrict__ aggs, const float* __restrict__ wsem,
                               float* __restrict__ out, int n, int M, long long slotStride,
                               const float* __restrict__ g, const float* __restrict__ bvec) {
  int lane = threadIdx.x & 31;
  int row = blockIdx.x * 8 + (threadIdx.x >> 5);
  if (row >= n) return;
  float4 v = {0.f, 0.f, 0.f, 0.f};
  for (int m = 0; m < M; ++m) {
    float wm = wsem[m];
    float4 a = ((const float4*)(aggs + (long long)m * slotStride + (size_t)row * 128))[lane];
    v.x += wm * fmaxf(a.x, 0.f);
    v.y += wm * fmaxf(a.y, 0.f);
    v.z += wm * fmaxf(a.z, 0.f);
    v.w += wm * fmaxf(a.w, 0.f);
  }
  v.x = v.x > 0.f ? v.x : (__expf(v.x) - 1.f);
  v.y = v.y > 0.f ? v.y : (__expf(v.y) - 1.f);
  v.z = v.z > 0.f ? v.z : (__expf(v.z) - 1.f);
  v.w = v.w > 0.f ? v.w : (__expf(v.w) - 1.f);
  float s = v.x + v.y + v.z + v.w;
#pragma unroll
  for (int o = 16; o; o >>= 1) s += __shfl_xor(s, o, 32);
  float mean = s * (1.f / 128.f);
  float dx = v.x - mean, dy = v.y - mean, dz = v.z - mean, dw = v.w - mean;
  float q = dx * dx + dy * dy + dz * dz + dw * dw;
#pragma unroll
  for (int o = 16; o; o >>= 1) q += __shfl_xor(q, o, 32);
  float rstd = rsqrtf(q * (1.f / 128.f) + 1e-5f);
  float4 gg = ((const float4*)g)[lane];
  float4 bb = ((const float4*)bvec)[lane];
  float4 r;
  r.x = dx * rstd * gg.x + bb.x;
  r.y = dy * rstd * gg.y + bb.y;
  r.z = dz * rstd * gg.z + bb.z;
  r.w = dw * rstd * gg.w + bb.w;
  ((float4*)(out + (size_t)row * 128))[lane] = r;
}

__global__ void pool_acc(const float* __restrict__ op, const int* __restrict__ batch,
                         float* __restrict__ ssum, float* __restrict__ cnt, int n) {
  int node = blockIdx.x, f = threadIdx.x;
  if (node >= n) return;
  int gidx = batch[node];
  atomicAdd(&ssum[(size_t)gidx * 128 + f], op[(size_t)node * 128 + f]);
  if (f == 0) atomicAdd(&cnt[gidx], 1.0f);
}

__global__ void final_lin(const float* __restrict__ ssum, const float* __restrict__ cnt,
                          const float* __restrict__ lW, const float* __restrict__ lb,
                          float* __restrict__ out, int G) {
  int lane = threadIdx.x & 31;
  int gidx = blockIdx.x * 8 + (threadIdx.x >> 5);
  if (gidx >= G) return;
  float c = cnt[gidx]; c = c > 1.f ? c : 1.f;
  float4 p = ((const float4*)(ssum + (size_t)gidx * 128))[lane];
  float4 w = ((const float4*)lW)[lane];
  float s = (p.x * w.x + p.y * w.y + p.z * w.z + p.w * w.w) / c;
#pragma unroll
  for (int o = 16; o; o >>= 1) s += __shfl_xor(s, o, 32);
  if (lane == 0) out[gidx] = s + lb[0];
}

// ---------------------------------------------------------------- host side

extern "C" void kernel_launch(void* const* d_in, const int* in_sizes, int n_in,
                              void* d_out, int out_size, void* d_ws, size_t ws_size,
                              hipStream_t stream) {
  (void)in_sizes; (void)n_in; (void)out_size; (void)ws_size;
  // node types: 0=operator 1=table 2=column 3=predicate
  const int Nn[4]    = {100000, 50000, 200000, 100000};
  const int indim[4] = {4, 2, 10, 1};
  const int Mnt[4]   = {4, 1, 1, 1};          // metapaths per dst type

  struct ETd { int si, di, E, st, dt, slot; };
  const ETd ets[7] = {
    { 4,  5, 200000, 1, 0, 0},   // table -> operator   (scannedby)
    { 6,  7, 200000, 3, 0, 1},   // predicate -> operator (filters)
    { 8,  9, 400000, 2, 0, 2},   // column -> operator  (outputby)
    {10, 11, 300000, 2, 3, 0},   // column -> predicate (connects)
    {12, 13, 200000, 0, 0, 3},   // operator -> operator (calledby)
    {14, 15,  50000, 1, 1, 0},   // table selfloop
    {16, 17, 200000, 2, 2, 0}};  // column selfloop

  // ---- workspace bump allocator (256B aligned) ----
  char* base = (char*)d_ws; size_t off = 0;
  auto alloc = [&](size_t bytes) -> void* {
    off = (off + 255) & ~(size_t)255;
    void* p = base + off; off += bytes; return p;
  };

  float* h[4]; float* xn[4];
  for (int t = 0; t < 4; ++t) h[t]  = (float*)alloc((size_t)Nn[t] * 128 * 4);
  for (int t = 0; t < 4; ++t) xn[t] = (float*)alloc((size_t)Nn[t] * 128 * 4);
  _Float16* packW = (_Float16*)alloc((size_t)10 * 16384 * 2);  // 10 weight mats

  size_t aggFloats = 0; size_t aggOffNT[4];
  for (int t = 0; t < 4; ++t) { aggOffNT[t] = aggFloats;
                                aggFloats += (size_t)Mnt[t] * Nn[t] * 128; }
  float* aggAll = (float*)alloc(aggFloats * 4);

  float *denom[7], *asrc[7], *adst[7], *exb[7];
  for (int e = 0; e < 7; ++e) {
    denom[e] = (float*)alloc((size_t)Nn[ets[e].dt] * 4);
    asrc[e]  = (float*)alloc((size_t)Nn[ets[e].st] * 4);
    adst[e]  = (float*)alloc((size_t)Nn[ets[e].dt] * 4);
    exb[e]   = (float*)alloc((size_t)ets[e].E * 4);
  }
  float* score = (float*)alloc(16 * 4);
  float* wsem  = (float*)alloc(16 * 4);
  float* ssum  = (float*)alloc((size_t)1024 * 128 * 4);
  float* cnt   = (float*)alloc(1024 * 4);

  // ---- pack all 128x128 weight matrices into WMMA B-fragment layout ----
  for (int t = 0; t < 4; ++t)
    pack_w<<<32, 32, 0, stream>>>((const float*)d_in[27 + 2 * t], packW + (size_t)t * 16384);
  pack_w<<<32, 32, 0, stream>>>((const float*)d_in[49], packW + (size_t)4 * 16384);
  for (int t = 0; t < 4; ++t)
    pack_w<<<32, 32, 0, stream>>>((const float*)d_in[52 + 2 * t], packW + (size_t)(5 + t) * 16384);
  pack_w<<<32, 32, 0, stream>>>((const float*)d_in[74], packW + (size_t)9 * 16384);

  // ---- encoders ----
  for (int t = 0; t < 4; ++t)
    encoder<<<Nn[t], 128, 0, stream>>>((const float*)d_in[t],
        (const float*)d_in[19 + 2 * t], (const float*)d_in[20 + 2 * t],
        h[t], Nn[t], indim[t]);

  // ---- one HANConv layer ----
  auto run_conv = [&](int projBase, int attSrcBase, int attDstBase,
                      int kbIdx, int qIdx, int packBase, int gIdx, int bIdx) {
    for (int t = 0; t < 4; ++t)
      gemm_bias<<<(Nn[t] + 127) / 128, 256, 0, stream>>>(h[t],
          packW + (size_t)(packBase + t) * 16384,
          (const float*)d_in[projBase + 2 * t + 1], xn[t], Nn[t]);

    zero_f32<<<8192, 256, 0, stream>>>(aggAll, (long long)aggFloats);
    for (int e = 0; e < 7; ++e)
      zero_f32<<<64, 256, 0, stream>>>(denom[e], Nn[ets[e].dt]);
    zero_f32<<<1, 32, 0, stream>>>(score, 16);

    // all attention dot-products, one pass over each xn[t]
    DotArgs da[4]; int dcnt[4] = {0, 0, 0, 0};
    for (int e = 0; e < 7; ++e) {
      const ETd& E = ets[e];
      da[E.st].v[dcnt[E.st]] = (const float*)d_in[attSrcBase + e];
      da[E.st].y[dcnt[E.st]++] = asrc[e];
      da[E.dt].v[dcnt[E.dt]] = (const float*)d_in[attDstBase + e];
      da[E.dt].y[dcnt[E.dt]++] = adst[e];
    }
    for (int t = 0; t < 4; ++t)
      rowdot_multi<<<(Nn[t] + 7) / 8, 256, 0, stream>>>(xn[t], da[t], dcnt[t], Nn[t]);

    for (int e = 0; e < 7; ++e) {
      const ETd& E = ets[e];
      edge_pass1<<<(E.E + 255) / 256, 256, 0, stream>>>(
          (const int*)d_in[E.si], (const int*)d_in[E.di],
          asrc[e], adst[e], exb[e], denom[e], E.E);
    }
    for (int e = 0; e < 7; ++e) {
      const ETd& E = ets[e];
      float* aggPtr = aggAll + aggOffNT[E.dt] + (size_t)E.slot * Nn[E.dt] * 128;
      edge_pass2<<<(E.E + 1) / 2, 256, 0, stream>>>(
          (const int*)d_in[E.si], (const int*)d_in[E.di],
          exb[e], denom[e], xn[E.st], aggPtr, E.E);
    }

    // semantic attention (relu fused into gemm_score A-load)
    for (int t = 0; t < 4; ++t)
      for (int m = 0; m < Mnt[t]; ++m)
        gemm_score<<<(Nn[t] + 127) / 128, 256, 0, stream>>>(
            aggAll + aggOffNT[t] + (size_t)m * Nn[t] * 128,
            packW + (size_t)(packBase + 4) * 16384,
            (const float*)d_in[kbIdx], (const float*)d_in[qIdx],
            score + t * 4 + m, Nn[t]);
    for (int t = 0; t < 4; ++t)
      softmax_w<<<1, 32, 0, stream>>>(score + t * 4, wsem + t * 4, Mnt[t],
                                      1.0f / (float)Nn[t]);
    // fused: weighted-combine(relu(agg)) + ELU + LayerNorm -> h
    for (int t = 0; t < 4; ++t)
      combine_elu_ln<<<(Nn[t] + 7) / 8, 256, 0, stream>>>(
          aggAll + aggOffNT[t], wsem + t * 4, h[t], Nn[t], Mnt[t],
          (long long)Nn[t] * 128,
          (const float*)d_in[gIdx], (const float*)d_in[bIdx]);
  };

  run_conv(27, 35, 42, 50, 51, 0, 77, 78);   // conv1 + norm1
  run_conv(52, 60, 67, 75, 76, 5, 79, 80);   // conv2 + norm2

  // ---- global mean pool over operator nodes + final linear ----
  zero_f32<<<512, 256, 0, stream>>>(ssum, (long long)1024 * 128);
  zero_f32<<<4, 256, 0, stream>>>(cnt, 1024);
  pool_acc<<<Nn[0], 128, 0, stream>>>(h[0], (const int*)d_in[18], ssum, cnt, Nn[0]);
  final_lin<<<(1024 + 7) / 8, 256, 0, stream>>>(ssum, cnt,
      (const float*)d_in[81], (const float*)d_in[82], (float*)d_out, 1024);
}
